// LSTMLayer_54614804136123
// MI455X (gfx1250) — compile-verified, matrix operand order checked by reading, not run on hardware
//
#include <hip/hip_runtime.h>
#include <hip/hip_bf16.h>

typedef __attribute__((ext_vector_type(16))) __bf16 v16bf;
typedef __attribute__((ext_vector_type(8)))  __bf16 v8bf;
typedef __attribute__((ext_vector_type(8)))  float  v8f;
typedef __attribute__((ext_vector_type(4)))  float  v4f;

#define T_DIM 4096
#define B_DIM 32
#define I_DIM 128
#define H_DIM 128
#define G_DIM 512                 // 4*H
#define ROWS  (T_DIM * B_DIM)     // 131072 gate rows (t*B + b)

#define BLOCK_THREADS 512         // 16 waves
#define ROWS_PER_WAVE 32          // 2 M-tiles
#define ROWS_PER_BLOCK (16 * ROWS_PER_WAVE)   // 512

// LDS layout (bytes)
#define SW_BYTES   (G_DIM * I_DIM * 2)        // 131072  bf16 W_ih [512][128]
#define SRB_BYTES  (B_DIM * G_DIM * 4)        //  65536  f32 rbias [32][512]
#define SC0_BYTES  (B_DIM * H_DIM * 4)        //  16384  f32 c0    [32][128]
#define SNZ_BYTES  (B_DIM * H_DIM * 4)        //  16384  f32 noise [32][128]
#define SMEM_BYTES (SW_BYTES + SRB_BYTES + SC0_BYTES + SNZ_BYTES)   // 229376

// ---------------- prep kernel 1: W_ih fp32 -> bf16 (L2-resident operand) ----
__global__ void cvt_w_kernel(const float* __restrict__ W, __bf16* __restrict__ Wb) {
    int i = blockIdx.x * 256 + threadIdx.x;     // 65536 elements total
    Wb[i] = (__bf16)W[i];
}

// ---------------- prep kernel 2: rbias[b][g] = h0[b,:]·W_hh[g,:] + b_ih + b_hh
__global__ void rbias_kernel(const float* __restrict__ h0,
                             const float* __restrict__ Whh,
                             const float* __restrict__ bih,
                             const float* __restrict__ bhh,
                             float* __restrict__ rb) {
    int i = blockIdx.x * 128 + threadIdx.x;     // 16384 outputs
    int b = i >> 9;
    int g = i & (G_DIM - 1);
    const float* hr = h0  + b * H_DIM;
    const float* wr = Whh + g * H_DIM;
    float acc = bih[g] + bhh[g];
#pragma unroll 4
    for (int k = 0; k < H_DIM; ++k) acc += hr[k] * wr[k];
    rb[i] = acc;
}

__device__ __forceinline__ float sigmoid_f(float x) {
    return 1.0f / (1.0f + __expf(-x));
}
__device__ __forceinline__ float tanh_f(float x) {
    return 1.0f - 2.0f / (__expf(2.0f * x) + 1.0f);
}

// ---------------- main fused GEMM (WMMA bf16) + LSTM activations ------------
// block = 512 threads = 16 waves; each wave owns 32 rows (2 M-tiles).
// W_ih (bf16), rbias, c0, noise staged in LDS once per block (amortized over
// 512 rows). Per 16-col group nt0, the 4 gate tiles (i,f,g,o) accumulate
// against the same A fragments, so the LSTM cell fuses on registers.
__global__ __launch_bounds__(BLOCK_THREADS) void lstm_main_kernel(
    const float*  __restrict__ X,      // [ROWS, 128]
    const __bf16* __restrict__ Wb,     // [512, 128] bf16 (g-major)
    const float*  __restrict__ rb,     // [32, 512]
    const float*  __restrict__ c0,     // [32, 128]
    const float*  __restrict__ noise,  // [32, 128]
    float*        __restrict__ out)    // [ROWS*128] ++ h_last[32*128] ++ c_last[32*128]
{
    extern __shared__ char smem[];
    __bf16* sW  = (__bf16*)smem;                                   // [512][128]
    float*  sRB = (float*)(smem + SW_BYTES);                       // [32][512]
    float*  sC0 = (float*)(smem + SW_BYTES + SRB_BYTES);           // [32][128]
    float*  sNZ = (float*)(smem + SW_BYTES + SRB_BYTES + SC0_BYTES);

    const int tid = threadIdx.x;

    // ---- cooperative stage of shared operands into LDS ----
    {
        const uint4* gw = (const uint4*)Wb;   uint4* sw = (uint4*)sW;
#pragma unroll
        for (int j = 0; j < 16; ++j)                       // 8192 x uint4
            sw[j * BLOCK_THREADS + tid] = gw[j * BLOCK_THREADS + tid];
        const float4* grb = (const float4*)rb;  float4* srb = (float4*)sRB;
#pragma unroll
        for (int j = 0; j < 8; ++j)                        // 4096 x float4
            srb[j * BLOCK_THREADS + tid] = grb[j * BLOCK_THREADS + tid];
        const float4* gc = (const float4*)c0;    float4* sc = (float4*)sC0;
        const float4* gn = (const float4*)noise; float4* sn = (float4*)sNZ;
#pragma unroll
        for (int j = 0; j < 2; ++j) {                      // 1024 x float4 each
            sc[j * BLOCK_THREADS + tid] = gc[j * BLOCK_THREADS + tid];
            sn[j * BLOCK_THREADS + tid] = gn[j * BLOCK_THREADS + tid];
        }
    }
    __syncthreads();

    const int lane = tid & 31;
    const int wave = tid >> 5;
    const int ln   = lane & 15;                 // N / M position within a tile
    const int kh   = lane >> 4;                 // lane-half -> K interleave group
    const int wRow = blockIdx.x * ROWS_PER_BLOCK + wave * ROWS_PER_WAVE;

    // Speculative prefetch of the next block's X rows (global_prefetch_b8).
    __builtin_prefetch(X + (size_t)(wRow + ln + ROWS_PER_BLOCK) * I_DIM, 0, 0);

    // ---- A fragments: 2 M-tiles x 4 K-steps (16x32 bf16 each).
    // ISA 16-bit A layout: lane-half kh holds K = 8*kh+[0..7] in v0..3 and
    // K = 16+8*kh+[0..7] in v4..7 (per 32-wide K step).
    v16bf afr[2][4];
#pragma unroll
    for (int mt = 0; mt < 2; ++mt) {
        const float* xr = X + (size_t)(wRow + mt * 16 + ln) * I_DIM;
#pragma unroll
        for (int kt = 0; kt < 4; ++kt) {
            const int k0 = kt * 32 + kh * 8;
            v4f x0 = *(const v4f*)(xr + k0);
            v4f x1 = *(const v4f*)(xr + k0 + 4);
            v4f x2 = *(const v4f*)(xr + k0 + 16);
            v4f x3 = *(const v4f*)(xr + k0 + 20);
            v16bf a;
#pragma unroll
            for (int j = 0; j < 4; ++j) {
                a[j]      = (__bf16)x0[j];
                a[4 + j]  = (__bf16)x1[j];
                a[8 + j]  = (__bf16)x2[j];
                a[12 + j] = (__bf16)x3[j];
            }
            afr[mt][kt] = a;
        }
    }

    // ---- 8 column groups; per group accumulate all 4 gate tiles (i,f,g,o).
    for (int nt0 = 0; nt0 < 8; ++nt0) {
        v8f acc[4][2] = {};     // [gate][mtile]

#pragma unroll
        for (int kt = 0; kt < 4; ++kt) {
#pragma unroll
            for (int g = 0; g < 4; ++g) {
                // B fragment from LDS: gate column g*128 + nt0*16 + ln
                const __bf16* wl = sW
                    + (size_t)(g * H_DIM + nt0 * 16 + ln) * I_DIM
                    + kt * 32 + kh * 8;
                v8bf b0 = *(const v8bf*)(wl);
                v8bf b1 = *(const v8bf*)(wl + 16);
                v16bf bfrag = __builtin_shufflevector(
                    b0, b1, 0,1,2,3,4,5,6,7,8,9,10,11,12,13,14,15);
#pragma unroll
                for (int mt = 0; mt < 2; ++mt) {
                    acc[g][mt] = __builtin_amdgcn_wmma_f32_16x16x32_bf16(
                        false, afr[mt][kt], false, bfrag,
                        (short)0, acc[g][mt], false, false);
                }
            }
        }

        // ---- fused LSTM cell on the accumulators.
        // C layout: element (VGPR v, lane) = (M = v + 8*kh, N = ln).
        const int h = nt0 * 16 + ln;
#pragma unroll
        for (int mt = 0; mt < 2; ++mt) {
#pragma unroll
            for (int v = 0; v < 8; ++v) {
                const int row = wRow + mt * 16 + v + 8 * kh;
                const int b   = row & (B_DIM - 1);
                const float* rrow = sRB + b * G_DIM;

                float gi = acc[0][mt][v] + rrow[h];
                float gf = acc[1][mt][v] + rrow[H_DIM + h];
                float gg = acc[2][mt][v] + rrow[2 * H_DIM + h];
                float go = acc[3][mt][v] + rrow[3 * H_DIM + h];

                float ig = sigmoid_f(gi);
                float fg = sigmoid_f(gf);
                float gt = tanh_f(gg);
                float og = sigmoid_f(go);

                float c  = fg * sC0[b * H_DIM + h] + ig * gt;
                float hv = og * tanh_f(c) + sNZ[b * H_DIM + h];

                out[(size_t)row * H_DIM + h] = hv;
                if (row >= ROWS - B_DIM) {   // t == T-1: emit (h_last, c_last)
                    out[(size_t)ROWS * H_DIM + b * H_DIM + h]                 = hv;
                    out[(size_t)ROWS * H_DIM + B_DIM * H_DIM + b * H_DIM + h] = c;
                }
            }
        }
    }
}

extern "C" void kernel_launch(void* const* d_in, const int* in_sizes, int n_in,
                              void* d_out, int out_size, void* d_ws, size_t ws_size,
                              hipStream_t stream) {
    const float* X     = (const float*)d_in[0];
    const float* h0    = (const float*)d_in[1];
    const float* c0    = (const float*)d_in[2];
    const float* noise = (const float*)d_in[3];
    const float* Wih   = (const float*)d_in[4];
    const float* Whh   = (const float*)d_in[5];
    const float* bih   = (const float*)d_in[6];
    const float* bhh   = (const float*)d_in[7];

    // workspace: [0, 128KB) bf16 W_ih ; [128KB, 192KB) f32 rbias[32][512]
    __bf16* Wb = (__bf16*)d_ws;
    float*  rb = (float*)((char*)d_ws + (size_t)SW_BYTES);

    cvt_w_kernel<<<(G_DIM * I_DIM) / 256, 256, 0, stream>>>(Wih, Wb);
    rbias_kernel<<<(B_DIM * G_DIM) / 128, 128, 0, stream>>>(h0, Whh, bih, bhh, rb);

    lstm_main_kernel<<<ROWS / ROWS_PER_BLOCK, BLOCK_THREADS, SMEM_BYTES, stream>>>(
        X, Wb, rb, c0, noise, (float*)d_out);
}
